// TransformerBlock_1219770712358
// MI455X (gfx1250) — compile-verified, matrix operand order checked by reading, not run on hardware
//
#include <hip/hip_runtime.h>

// ---------------- types ----------------
typedef __bf16 bf16x16 __attribute__((ext_vector_type(16)));
typedef float  v8f     __attribute__((ext_vector_type(8)));
typedef unsigned int u32x4 __attribute__((ext_vector_type(4)));
typedef int i32x4 __attribute__((ext_vector_type(4)));
typedef int i32x8 __attribute__((ext_vector_type(8)));

struct alignas(16) U4 { unsigned int x, y, z, w; };
struct alignas(16) F4 { float x, y, z, w; };

union Frag {
  U4 q[2];        // 2 x 16B = 8 VGPRs of packed bf16
  bf16x16 v;      // WMMA operand view
};

__device__ __forceinline__ unsigned short f32_to_bf16(float f) {
  union { float f; unsigned int u; } c; c.f = f;
  unsigned int r = c.u + 0x7FFFu + ((c.u >> 16) & 1u);   // round-to-nearest-even
  return (unsigned short)(r >> 16);
}
__device__ __forceinline__ float bf16_to_f32(unsigned short h) {
  union { unsigned int u; float f; } c; c.u = ((unsigned int)h) << 16;
  return c.f;
}

// TDM availability (tensor_load_to_lds builtin; arity differs by toolchain)
#if defined(__has_builtin)
#if __has_builtin(__builtin_amdgcn_tensor_load_to_lds) && __has_builtin(__builtin_amdgcn_s_wait_tensorcnt)
#define HAVE_TDM 1
#endif
#endif
#ifndef HAVE_TDM
#define HAVE_TDM 0
#endif

// model dims (fixed by reference)
#define BB 2
#define LL 2048
#define DD 1024
#define HH 16
#define DH 64
#define DFF 4096
#define TD 3072   // 3*D

// ---------------- RMSNorm (fp32 in -> bf16 out) ----------------
__global__ __launch_bounds__(256)
void rmsnorm_bf16_kernel(const float* __restrict__ x,
                         const float* __restrict__ g,
                         unsigned short* __restrict__ y) {
  __shared__ float red[256];
  const int row = blockIdx.x;
  const int tid = threadIdx.x;
  const float* xr = x + (size_t)row * DD;
  float v0 = xr[tid], v1 = xr[tid + 256], v2 = xr[tid + 512], v3 = xr[tid + 768];
  red[tid] = v0*v0 + v1*v1 + v2*v2 + v3*v3;
  __syncthreads();
  for (int s = 128; s > 0; s >>= 1) {
    if (tid < s) red[tid] += red[tid + s];
    __syncthreads();
  }
  const float inv = rsqrtf(red[0] * (1.0f / (float)DD) + 1e-5f);
  unsigned short* yr = y + (size_t)row * DD;
  yr[tid]       = f32_to_bf16(v0 * inv * g[tid]);
  yr[tid + 256] = f32_to_bf16(v1 * inv * g[tid + 256]);
  yr[tid + 512] = f32_to_bf16(v2 * inv * g[tid + 512]);
  yr[tid + 768] = f32_to_bf16(v3 * inv * g[tid + 768]);
}

// ---------------- RoPE (interleaved pairs) in place on q,k of qkv ----------------
__global__ __launch_bounds__(256)
void rope_kernel(unsigned short* __restrict__ qkv) {
  const int row = blockIdx.x;             // b*L + pos
  const int pos = row & (LL - 1);
  unsigned short* r = qkv + (size_t)row * TD;
  const float kln = 0.2878231366f;        // ln(10000)/32
  #pragma unroll
  for (int i = 0; i < 4; ++i) {
    int pi   = threadIdx.x + i * 256;     // 0..1023 pair index over q|k
    int qk   = pi >> 9;                   // 0 = q, 1 = k
    int idx  = pi & 511;
    int head = idx >> 5;
    int p    = idx & 31;
    int col  = qk * DD + head * DH + p * 2;
    float ang = (float)pos * __expf(-(float)p * kln);
    float s, c;
    __sincosf(ang, &s, &c);
    float x0 = bf16_to_f32(r[col]);
    float x1 = bf16_to_f32(r[col + 1]);
    r[col]     = f32_to_bf16(x0 * c - x1 * s);
    r[col + 1] = f32_to_bf16(x1 * c + x0 * s);
  }
}

// ---------------- bf16 WMMA GEMM:  C[m,n] = sum_k A[m,k] * W[n,k] ----------------
// Block tile 128(M) x 128(N) x 32(K); 8 waves, each 32x64 (2x4 WMMA tiles).
// A tile staged LDS<-global by the Tensor Data Mover when available.
// MODE 0: outb = bf16(C)
// MODE 1: outb = bf16(silu(C))
// MODE 2: outb = bf16(outb * C)
// MODE 3: outf = res + C        (fp32)
template<int MODE>
__global__ __launch_bounds__(256)
void gemm_bf16_wmma(const unsigned short* __restrict__ A,
                    const float* __restrict__ W,
                    int K, int N,
                    unsigned short* __restrict__ outb,
                    float* __restrict__ outf,
                    const float* __restrict__ res) {
  __shared__ unsigned short As[128 * 32];
  __shared__ unsigned short Ws[128 * 32];
  const int tid  = threadIdx.x;
  const int lane = tid & 31;
  const int wid  = tid >> 5;
  const int wm = (wid & 3) * 32;          // 4 waves along M, 32 rows each
  const int wn = (wid >> 2) * 64;         // 2 waves along N, 64 cols each
  const int blockM = blockIdx.y * 128;
  const int blockN = blockIdx.x * 128;
  const int Mtot = (int)gridDim.y * 128;

  const v8f vzero = {0.f,0.f,0.f,0.f,0.f,0.f,0.f,0.f};
  v8f acc[2][4];
  #pragma unroll
  for (int i = 0; i < 2; ++i)
    #pragma unroll
    for (int j = 0; j < 4; ++j) acc[i][j] = vzero;

  const int arow = tid >> 1;              // 0..127 (fallback A staging)
  const int acol = (tid & 1) * 16;        // 0 or 16
  const unsigned short* Aptr = A + (size_t)(blockM + arow) * K + acol;
  const unsigned short* Abase = A + (size_t)blockM * K;

  const int lm = lane & 15;               // row within 16 (A: M row, B: N row)
  const int kh = (lane >> 4) * 8;         // K sub-chunk per lane half (ISA layout)
#if HAVE_TDM
  const unsigned ldsAs = (unsigned)(size_t)(void*)As;   // LDS byte offset of As
#endif

  for (int kk = 0; kk < K; kk += 32) {
    __syncthreads();
#if HAVE_TDM
    // ---- A tile via Tensor Data Mover: 2D tile 32(x) x 128(y), bf16, stride K ----
    if (wid == 0) {
      unsigned long long ga = (unsigned long long)(Abase + kk);   // byte address of tile start
      u32x4 g0 = { 1u,                                            // count=1, user descriptor
                   ldsAs,                                         // lds_addr [63:32]
                   (unsigned)ga,                                  // global_addr [95:64]
                   (unsigned)((ga >> 32) & 0x1FFFFFFu) | (2u << 30) }; // addr[56:32] | type=2
      i32x8 g1 = { (int)0x00010000u,                              // wg_mask=0, data_size=1 (2B)
                   (int)((unsigned)(K & 0xFFFF) << 16),           // tensor_dim0 lo16 in [31:16]
                   (int)(((unsigned)K >> 16) | ((unsigned)(Mtot & 0xFFFF) << 16)), // dim0 hi | dim1 lo
                   (int)(((unsigned)Mtot >> 16) | (32u << 16)),   // dim1 hi | tile_dim0=32
                   (int)128,                                      // tile_dim1=128, tile_dim2=0
                   (int)K,                                        // tensor_dim0_stride lo32
                   0, 0 };                                        // stride hi, dim1_stride=0
      i32x4 gz = {0, 0, 0, 0};
#if __clang_major__ >= 23
      i32x8 z8 = {0,0,0,0,0,0,0,0};
      __builtin_amdgcn_tensor_load_to_lds(g0, g1, gz, gz, z8, 0);
#else
      __builtin_amdgcn_tensor_load_to_lds(g0, g1, gz, gz, 0);
#endif
      __builtin_amdgcn_s_wait_tensorcnt(0);
    }
#else
    // ---- fallback: manual A tile staging (128x32 bf16) ----
    {
      const U4* src = (const U4*)(Aptr + kk);
      U4 a0 = src[0], a1 = src[1];
      U4* dst = (U4*)&As[arow * 32 + acol];
      dst[0] = a0; dst[1] = a1;
    }
    if (kk + 32 < K) __builtin_prefetch(Aptr + kk + 32, 0, 1);
#endif
    // ---- W tile (128x32), fp32 -> bf16 on the fly ----
    #pragma unroll
    for (int t = 0; t < 4; ++t) {
      int li  = tid * 4 + t;              // 0..1023
      int row = li >> 3;                  // 0..127
      int c0  = (li & 7) * 4;             // 0..28
      F4 w = *(const F4*)(W + (size_t)(blockN + row) * K + kk + c0);
      unsigned int p0 = (unsigned int)f32_to_bf16(w.x) | ((unsigned int)f32_to_bf16(w.y) << 16);
      unsigned int p1 = (unsigned int)f32_to_bf16(w.z) | ((unsigned int)f32_to_bf16(w.w) << 16);
      unsigned int* dst = (unsigned int*)&Ws[row * 32 + c0];
      dst[0] = p0; dst[1] = p1;
    }
    __syncthreads();

    // ---- fragments first, then the WMMA burst ----
    Frag af[2], bfr[4];
    #pragma unroll
    for (int i = 0; i < 2; ++i) {
      const unsigned short* p = &As[(wm + i * 16 + lm) * 32 + kh];
      af[i].q[0] = *(const U4*)p;
      af[i].q[1] = *(const U4*)(p + 16);
    }
    #pragma unroll
    for (int j = 0; j < 4; ++j) {
      const unsigned short* p = &Ws[(wn + j * 16 + lm) * 32 + kh];
      bfr[j].q[0] = *(const U4*)p;
      bfr[j].q[1] = *(const U4*)(p + 16);
    }
    #pragma unroll
    for (int i = 0; i < 2; ++i)
      #pragma unroll
      for (int j = 0; j < 4; ++j)
        acc[i][j] = __builtin_amdgcn_wmma_f32_16x16x32_bf16(
            false, af[i].v, false, bfr[j].v, (short)0, acc[i][j], false, false);
  }

  // epilogue (C/D layout: lane = N col, VGPR r = M row within 8-row half)
  const int nn = lane & 15;
  const int mb = (lane >> 4) * 8;
  #pragma unroll
  for (int i = 0; i < 2; ++i) {
    #pragma unroll
    for (int j = 0; j < 4; ++j) {
      const int tm = blockM + wm + i * 16 + mb;
      const int tn = blockN + wn + j * 16 + nn;
      #pragma unroll
      for (int r = 0; r < 8; ++r) {
        float v = acc[i][j][r];
        size_t idx = (size_t)(tm + r) * N + tn;
        if (MODE == 0) {
          outb[idx] = f32_to_bf16(v);
        } else if (MODE == 1) {
          outb[idx] = f32_to_bf16(v / (1.0f + __expf(-v)));
        } else if (MODE == 2) {
          float prev = bf16_to_f32(outb[idx]);
          outb[idx] = f32_to_bf16(prev * v);
        } else {
          outf[idx] = res[idx] + v;
        }
      }
    }
  }
}

// ---------------- flash attention (causal, bf16 WMMA, online softmax) ----------------
__global__ __launch_bounds__(128)
void attn_kernel(const unsigned short* __restrict__ qkv,
                 unsigned short* __restrict__ attb) {
  __shared__ unsigned short Ks[32 * 64];       // [key][dh]
  __shared__ unsigned short Vt[64 * 32];       // [dh][key]  (transposed for B-operand reads)
  __shared__ unsigned short Pb[4 * 16 * 32];   // per-wave P tile (C->A relayout)
  const int tid  = threadIdx.x;
  const int lane = tid & 31;
  const int wid  = tid >> 5;
  const int b = blockIdx.y >> 4;
  const int h = blockIdx.y & 15;
  const int q0 = blockIdx.x * 64;
  const int qw = q0 + wid * 16;                // this wave's 16 query rows
  const int lm = lane & 15;
  const int kh = (lane >> 4) * 8;
  const int nn = lm;
  const int mb = (lane >> 4) * 8;

  // Q fragments: A-operand, two dh chunks of 32 (DH=64)
  const unsigned short* qbase = qkv + (size_t)(b * LL + qw + lm) * TD + h * DH;
  Frag qf[2];
  #pragma unroll
  for (int c = 0; c < 2; ++c) {
    qf[c].q[0] = *(const U4*)(qbase + c * 32 + kh);
    qf[c].q[1] = *(const U4*)(qbase + c * 32 + kh + 16);
  }

  const v8f vzero = {0.f,0.f,0.f,0.f,0.f,0.f,0.f,0.f};
  float rm[8], rl[8];
  v8f o[4];
  #pragma unroll
  for (int r = 0; r < 8; ++r) { rm[r] = -1e30f; rl[r] = 0.f; }
  #pragma unroll
  for (int t = 0; t < 4; ++t) o[t] = vzero;

  const int nkb = q0 / 32 + 2;                 // key blocks covering keys 0..q0+63
  const unsigned short* kvbase = qkv + (size_t)(b * LL) * TD;
  unsigned short* Pw = &Pb[wid * 16 * 32];

  for (int kb = 0; kb < nkb; ++kb) {
    __syncthreads();
    // cooperative load: K block (rows as-is), V block (transposed into Vt)
    {
      const unsigned short* kbp = kvbase + (size_t)(kb * 32) * TD + DD + h * DH;
      const unsigned short* vbp = kvbase + (size_t)(kb * 32) * TD + 2 * DD + h * DH;
      #pragma unroll
      for (int t = 0; t < 2; ++t) {
        int li  = tid * 2 + t;                 // 0..255
        int row = li >> 3;                     // 0..31 (key within block)
        int c0  = (li & 7) * 8;                // 0..56 (dh)
        U4 kq = *(const U4*)(kbp + (size_t)row * TD + c0);
        *(U4*)&Ks[row * 64 + c0] = kq;
        U4 vq = *(const U4*)(vbp + (size_t)row * TD + c0);
        const unsigned short* vs = (const unsigned short*)&vq;
        #pragma unroll
        for (int e = 0; e < 8; ++e) Vt[(c0 + e) * 32 + row] = vs[e];
      }
    }
    __syncthreads();

    // S = Q K^T  (two 16x16 n-tiles over 32 keys, 2 k-steps over dh)
    v8f s[2] = {vzero, vzero};
    #pragma unroll
    for (int c = 0; c < 2; ++c)
      #pragma unroll
      for (int j = 0; j < 2; ++j) {
        Frag kf;
        const unsigned short* p = &Ks[(j * 16 + lm) * 64 + c * 32 + kh];
        kf.q[0] = *(const U4*)p;
        kf.q[1] = *(const U4*)(p + 16);
        s[j] = __builtin_amdgcn_wmma_f32_16x16x32_bf16(
            false, qf[c].v, false, kf.v, (short)0, s[j], false, false);
      }

    // scale + causal mask
    float sv[2][8];
    #pragma unroll
    for (int j = 0; j < 2; ++j) {
      int key = kb * 32 + j * 16 + nn;
      #pragma unroll
      for (int r = 0; r < 8; ++r) {
        float v = s[j][r] * 0.125f;            // 1/sqrt(64)
        int qrow = qw + mb + r;
        sv[j][r] = (key > qrow) ? -1e30f : v;
      }
    }
    // online softmax: row reductions across the 16-lane half holding each row
    #pragma unroll
    for (int r = 0; r < 8; ++r) {
      float bm = fmaxf(sv[0][r], sv[1][r]);
      #pragma unroll
      for (int off = 8; off > 0; off >>= 1) bm = fmaxf(bm, __shfl_xor(bm, off, 16));
      float newm = fmaxf(rm[r], bm);
      float sc = __expf(rm[r] - newm);
      float p0 = __expf(sv[0][r] - newm);
      float p1 = __expf(sv[1][r] - newm);
      float ps = p0 + p1;
      #pragma unroll
      for (int off = 8; off > 0; off >>= 1) ps += __shfl_xor(ps, off, 16);
      rl[r] = rl[r] * sc + ps;
      rm[r] = newm;
      #pragma unroll
      for (int t = 0; t < 4; ++t) o[t][r] *= sc;
      Pw[(mb + r) * 32 + nn]      = f32_to_bf16(p0);   // C->A relayout via LDS
      Pw[(mb + r) * 32 + 16 + nn] = f32_to_bf16(p1);
    }

    // O += P * V   (K=32 keys = one WMMA k-step; 4 n-tiles over dh)
    Frag pf;
    {
      const unsigned short* p = &Pw[lm * 32 + kh];
      pf.q[0] = *(const U4*)p;
      pf.q[1] = *(const U4*)(p + 16);
    }
    #pragma unroll
    for (int t = 0; t < 4; ++t) {
      Frag vf;
      const unsigned short* p = &Vt[(t * 16 + lm) * 32 + kh];
      vf.q[0] = *(const U4*)p;
      vf.q[1] = *(const U4*)(p + 16);
      o[t] = __builtin_amdgcn_wmma_f32_16x16x32_bf16(
          false, pf.v, false, vf.v, (short)0, o[t], false, false);
    }
  }

  // normalize and store bf16 head slice
  #pragma unroll
  for (int t = 0; t < 4; ++t)
    #pragma unroll
    for (int r = 0; r < 8; ++r) {
      int qrow = qw + mb + r;
      float v = o[t][r] / rl[r];
      attb[(size_t)(b * LL + qrow) * DD + h * DH + t * 16 + nn] = f32_to_bf16(v);
    }
}

// ---------------- launch ----------------
extern "C" void kernel_launch(void* const* d_in, const int* in_sizes, int n_in,
                              void* d_out, int out_size, void* d_ws, size_t ws_size,
                              hipStream_t stream) {
  (void)in_sizes; (void)n_in; (void)out_size;
  const float* x     = (const float*)d_in[0];
  const float* w_qkv = (const float*)d_in[1];
  const float* w_o   = (const float*)d_in[2];
  const float* w1    = (const float*)d_in[3];
  const float* w2    = (const float*)d_in[4];
  const float* w3    = (const float*)d_in[5];
  const float* g1    = (const float*)d_in[6];
  const float* g2    = (const float*)d_in[7];
  float* out = (float*)d_out;

  const size_t NT = (size_t)BB * LL;                    // 4096 tokens
  char* p = (char*)d_ws;
  unsigned short* yb   = (unsigned short*)p; p += NT * DD  * 2;  //  8 MiB
  unsigned short* qkvb = (unsigned short*)p; p += NT * TD  * 2;  // 24 MiB
  unsigned short* attb = (unsigned short*)p; p += NT * DD  * 2;  //  8 MiB
  float*          x2   = (float*)p;          p += NT * DD  * 4;  // 16 MiB
  unsigned short* y2b  = (unsigned short*)p; p += NT * DD  * 2;  //  8 MiB
  unsigned short* hb   = (unsigned short*)p; p += NT * DFF * 2;  // 32 MiB
  if ((size_t)(p - (char*)d_ws) > ws_size) return;               // ws too small: bail

  // attention sublayer
  rmsnorm_bf16_kernel<<<NT, 256, 0, stream>>>(x, g1, yb);
  gemm_bf16_wmma<0><<<dim3(TD / 128, NT / 128), 256, 0, stream>>>(yb, w_qkv, DD, TD, qkvb, nullptr, nullptr);
  rope_kernel<<<NT, 256, 0, stream>>>(qkvb);
  attn_kernel<<<dim3(LL / 64, BB * HH), 128, 0, stream>>>(qkvb, attb);
  gemm_bf16_wmma<3><<<dim3(DD / 128, NT / 128), 256, 0, stream>>>(attb, w_o, DD, DD, nullptr, x2, x);

  // SwiGLU FFN sublayer
  rmsnorm_bf16_kernel<<<NT, 256, 0, stream>>>(x2, g2, y2b);
  gemm_bf16_wmma<1><<<dim3(DFF / 128, NT / 128), 256, 0, stream>>>(y2b, w1, DD, DFF, hb, nullptr, nullptr);
  gemm_bf16_wmma<2><<<dim3(DFF / 128, NT / 128), 256, 0, stream>>>(y2b, w3, DD, DFF, hb, nullptr, nullptr);
  gemm_bf16_wmma<3><<<dim3(DD / 128, NT / 128), 256, 0, stream>>>(hb, w2, DFF, DD, nullptr, out, x2);
}
